// EpipolarReward_7593502179528
// MI455X (gfx1250) — matrix operand order repaired and checked by reading, not compiled, not run on hardware
//
#include <hip/hip_runtime.h>
#include <hip/hip_fp16.h>
#include <hip/hip_bf16.h>

typedef _Float16 half_t;
typedef __attribute__((ext_vector_type(16))) _Float16 v16h;
typedef __attribute__((ext_vector_type(8)))  _Float16 v8h;
typedef __attribute__((ext_vector_type(4)))  _Float16 v4h;
typedef __attribute__((ext_vector_type(8)))  float    v8f;

#define P_    120
#define N_    1024
#define M_    1024
#define D_    256
#define TILE  128
#define NB    (N_/TILE)   // 8 row tiles
#define MB    (M_/TILE)   // 8 col tiles
#define EPSC  1e-6f
#define THRESH 0.8f

// ---------------------------------------------------------------------------
// Stage A: L2-normalize fp32 descriptor rows -> f16 rows in workspace.
// One wave (32 lanes) per 256-element row; lane handles floats 4*lane..+3 and 128+4*lane..+3.
// ---------------------------------------------------------------------------
__global__ __launch_bounds__(256)
void ep_normalize_kernel(const float* __restrict__ src, half_t* __restrict__ dst, int rows) {
  int gt   = blockIdx.x * blockDim.x + threadIdx.x;
  int wave = gt >> 5;
  int lane = threadIdx.x & 31;
  if (wave >= rows) return;
  const float4* r = (const float4*)(src + (size_t)wave * D_);
  float4 a = r[lane];        // floats 4*lane     .. 4*lane+3
  float4 b = r[lane + 32];   // floats 128+4*lane .. 128+4*lane+3
  float s = a.x*a.x + a.y*a.y + a.z*a.z + a.w*a.w
          + b.x*b.x + b.y*b.y + b.z*b.z + b.w*b.w;
  #pragma unroll
  for (int m = 16; m; m >>= 1) s += __shfl_xor(s, m, 32);
  float inv = rsqrtf(s + 1e-12f);
  half_t* o = dst + (size_t)wave * D_;
  v4h ha, hb;
  ha[0] = (half_t)(a.x*inv); ha[1] = (half_t)(a.y*inv);
  ha[2] = (half_t)(a.z*inv); ha[3] = (half_t)(a.w*inv);
  hb[0] = (half_t)(b.x*inv); hb[1] = (half_t)(b.y*inv);
  hb[2] = (half_t)(b.z*inv); hb[3] = (half_t)(b.w*inv);
  *(v4h*)(o + 4*lane)        = ha;
  *(v4h*)(o + 128 + 4*lane)  = hb;
}

// ---------------------------------------------------------------------------
// Stage B: per 128x128 tile: f16 WMMA GEMM of normalized descriptors, then
// per-row / per-column top-2 partials selected on the RAW dot product s
// (sqrt(max(2-2s,0)) is monotone-decreasing in s, so max-s == min-dist and
// ties break the same way). Distances materialize only in Stage C.
// block = 256 threads (8 waves); wave w owns rows [16w,16w+16) x 128 cols.
// ---------------------------------------------------------------------------
__global__ __launch_bounds__(256)
void ep_gemm_top2_kernel(const half_t* __restrict__ nI, const half_t* __restrict__ nJ,
                         float4* __restrict__ rowPart, float4* __restrict__ colPart) {
  const int cb = blockIdx.x, rb = blockIdx.y, p = blockIdx.z;
  const int tid   = threadIdx.x;
  const int w     = tid >> 5;
  const int lane  = tid & 31;
  const int lhalf = lane >> 4;     // 0: lanes 0-15, 1: lanes 16-31
  const int l15   = lane & 15;

  __shared__ float sdot[TILE][TILE + 1];   // stride 129 floats: conflict-free both scans

  // A fragment source: row = rb*128 + w*16 + l15
  // A layout (16-bit A 16x32): lanes 0-15 hold K k0..k0+15,
  //                            lanes 16-31 hold K k0+16..k0+31
  const half_t* Abase = nI + ((size_t)p * N_ + (size_t)rb * TILE + w * 16 + l15) * D_;
  // B fragment source: col n = cb*128 + t*16 + l15 maps to desc_j row n.
  // Layout used here (B 32x16, same k order as A): lanes 0-15 hold K k0..k0+15, lanes 16-31 hold K k0+16..k0+31
  const half_t* Bbase = nJ + ((size_t)p * M_ + (size_t)cb * TILE + l15) * D_;

  v8f acc[8];
  #pragma unroll
  for (int t = 0; t < 8; ++t) acc[t] = v8f{};

  for (int k0 = 0; k0 < D_; k0 += 32) {
    union { v16h v; v8h h[2]; } af;
    af.h[0] = *(const v8h*)(Abase + k0 + lhalf * 16);
    af.h[1] = *(const v8h*)(Abase + k0 + 8 + lhalf * 16);
    #pragma unroll
    for (int t = 0; t < 8; ++t) {
      v16h bf = *(const v16h*)(Bbase + (size_t)t * 16 * D_ + k0 + lhalf * 16);
      acc[t] = __builtin_amdgcn_wmma_f32_16x16x32_f16(
          false, af.v, false, bf, (short)0, acc[t], false, false);
    }
  }

  // C layout: acc[t][r] at lane l = S[row = r + (l>=16 ? 8:0)][col = l&15]
  #pragma unroll
  for (int t = 0; t < 8; ++t) {
    #pragma unroll
    for (int r = 0; r < 8; ++r) {
      sdot[w * 16 + r + lhalf * 8][t * 16 + l15] = acc[t][r];
    }
  }
  __syncthreads();

  // ---- per-row top-2 (max s) over this tile's 128 columns (2 threads/row) ----
  {
    int row = tid >> 1, h = tid & 1;
    float v0 = -1e30f, v1 = -1e30f; int i0 = 0;
    const float* rp = &sdot[row][h * 64];
    #pragma unroll 4
    for (int c = 0; c < 64; ++c) {
      float s  = rp[c];
      float o0 = v0;
      v1 = fmaxf(v1, fminf(o0, s));   // second-best = max(v1, min(v0, s))
      v0 = fmaxf(o0, s);
      i0 = (s > o0) ? (h * 64 + c) : i0;
    }
    float ov0 = __shfl_xor(v0, 1, 32);
    float ov1 = __shfl_xor(v1, 1, 32);
    int   oi0 = __shfl_xor(i0, 1, 32);
    if (h == 0) {   // merge, lower-index half wins ties (strict >)
      float nv0, nv1; int ni0;
      if (ov0 > v0) { nv0 = ov0; ni0 = oi0; nv1 = fmaxf(v0, ov1); }
      else          { nv0 = v0;  ni0 = i0;  nv1 = fmaxf(v1, ov0); }
      float4 o; o.x = nv0; o.y = nv1;
      o.z = __int_as_float(cb * TILE + ni0); o.w = 0.0f;
      rowPart[(((size_t)p * NB + rb) * MB + cb) * TILE + row] = o;
    }
  }
  // ---- per-column top-2 (max s) over this tile's 128 rows (2 threads/col) ----
  {
    int col = tid >> 1, h = tid & 1;
    float v0 = -1e30f, v1 = -1e30f; int i0 = 0;
    #pragma unroll 4
    for (int r = 0; r < 64; ++r) {
      float s  = sdot[h * 64 + r][col];
      float o0 = v0;
      v1 = fmaxf(v1, fminf(o0, s));
      v0 = fmaxf(o0, s);
      i0 = (s > o0) ? (h * 64 + r) : i0;
    }
    float ov0 = __shfl_xor(v0, 1, 32);
    float ov1 = __shfl_xor(v1, 1, 32);
    int   oi0 = __shfl_xor(i0, 1, 32);
    if (h == 0) {
      float nv0, nv1; int ni0;
      if (ov0 > v0) { nv0 = ov0; ni0 = oi0; nv1 = fmaxf(v0, ov1); }
      else          { nv0 = v0;  ni0 = i0;  nv1 = fmaxf(v1, ov0); }
      float4 o; o.x = nv0; o.y = nv1;
      o.z = __int_as_float(rb * TILE + ni0); o.w = 0.0f;
      colPart[(((size_t)p * MB + cb) * NB + rb) * TILE + col] = o;
    }
  }
}

// ---------------------------------------------------------------------------
// Stage C: fold 8 tile-partials per line into final top-2 (still in s-space),
// then convert the two winners to true distances and apply the ratio test.
// Works for both row direction (part idx [p][rb][cb][r]) and column direction.
// ---------------------------------------------------------------------------
__global__ __launch_bounds__(256)
void ep_merge_top2_kernel(const float4* __restrict__ part, float4* __restrict__ best, int total) {
  int idx = blockIdx.x * blockDim.x + threadIdx.x;
  if (idx >= total) return;
  int p     = idx >> 10;
  int line  = idx & 1023;
  int outer = line >> 7, local = line & 127;
  const float4* q = part + ((size_t)(p * 8 + outer) * 8) * TILE + local;
  float v0 = -1e30f, v1 = -1e30f; int i0 = 0;
  #pragma unroll
  for (int c = 0; c < 8; ++c) {
    float4 e = q[(size_t)c * TILE];
    if (e.x > v0) { v1 = fmaxf(v0, e.y); v0 = e.x; i0 = __float_as_int(e.z); }
    else          { v1 = fmaxf(v1, e.x); }
  }
  // s -> distance (monotone decreasing): d = sqrt(max(2-2s, 0))
  float d0 = sqrtf(fmaxf(2.0f - 2.0f * v0, 0.0f));
  float d1 = sqrtf(fmaxf(2.0f - 2.0f * v1, 0.0f));
  float ratio = d0 / (d1 + EPSC);
  float4 o; o.x = d0; o.y = d1; o.z = __int_as_float(i0);
  o.w = (ratio < THRESH) ? 1.0f : 0.0f;
  best[idx] = o;
}

// ---------------------------------------------------------------------------
// Stage D: mutual check + Sampson error, block-reduced, atomics into accum.
// ---------------------------------------------------------------------------
__global__ __launch_bounds__(256)
void ep_sampson_kernel(const float4* __restrict__ rowBest, const float4* __restrict__ colBest,
                       const float* __restrict__ kp_i, const float* __restrict__ kp_j,
                       const float* __restrict__ F, float* __restrict__ accum) {
  int idx = blockIdx.x * blockDim.x + threadIdx.x;   // p*N + i
  float err = 0.0f, cnt = 0.0f;
  if (idx < P_ * N_) {
    int p = idx >> 10, i = idx & 1023;
    float4 rbv = rowBest[idx];
    if (rbv.w > 0.5f) {
      int j0 = __float_as_int(rbv.z);
      float4 cbv = colBest[(size_t)p * M_ + j0];
      if (cbv.w > 0.5f && __float_as_int(cbv.z) == i) {
        const float* Fp = F + (size_t)p * 9;
        float xi = kp_i[((size_t)p * N_ + i)  * 2], yi = kp_i[((size_t)p * N_ + i)  * 2 + 1];
        float xj = kp_j[((size_t)p * M_ + j0) * 2], yj = kp_j[((size_t)p * M_ + j0) * 2 + 1];
        float lj0 = Fp[0]*xi + Fp[1]*yi + Fp[2];
        float lj1 = Fp[3]*xi + Fp[4]*yi + Fp[5];
        float lj2 = Fp[6]*xi + Fp[7]*yi + Fp[8];
        float li0 = Fp[0]*xj + Fp[3]*yj + Fp[6];
        float li1 = Fp[1]*xj + Fp[4]*yj + Fp[7];
        float num = xj*lj0 + yj*lj1 + lj2;  num *= num;
        float den = lj0*lj0 + lj1*lj1 + li0*li0 + li1*li1 + EPSC;
        err = num / den; cnt = 1.0f;
      }
    }
  }
  __shared__ float se[256], sc[256];
  se[threadIdx.x] = err; sc[threadIdx.x] = cnt;
  __syncthreads();
  #pragma unroll
  for (int s = 128; s; s >>= 1) {
    if (threadIdx.x < s) { se[threadIdx.x] += se[threadIdx.x + s]; sc[threadIdx.x] += sc[threadIdx.x + s]; }
    __syncthreads();
  }
  if (threadIdx.x == 0) { atomicAdd(accum, se[0]); atomicAdd(accum + 1, sc[0]); }
}

__global__ void ep_zero_accum_kernel(float* a) { if (threadIdx.x < 2) a[threadIdx.x] = 0.0f; }

__global__ void ep_finalize_kernel(const float* __restrict__ accum, float* __restrict__ out) {
  float e = accum[0], c = accum[1];
  float mean = e / fmaxf(c, 1.0f);
  out[0] = expf(-mean);
  out[1] = mean;
  out[2] = c;
}

// ---------------------------------------------------------------------------
extern "C" void kernel_launch(void* const* d_in, const int* in_sizes, int n_in,
                              void* d_out, int out_size, void* d_ws, size_t ws_size,
                              hipStream_t stream) {
  (void)in_sizes; (void)n_in; (void)out_size; (void)ws_size;
  const float* desc_i = (const float*)d_in[0];
  const float* desc_j = (const float*)d_in[1];
  const float* kp_i   = (const float*)d_in[2];
  const float* kp_j   = (const float*)d_in[3];
  const float* F      = (const float*)d_in[4];

  char* ws = (char*)d_ws;
  size_t offNI = 0;
  size_t offNJ = offNI + (size_t)P_ * N_ * D_ * sizeof(half_t);
  size_t offRP = offNJ + (size_t)P_ * M_ * D_ * sizeof(half_t);
  size_t offCP = offRP + (size_t)P_ * NB * MB * TILE * sizeof(float4);
  size_t offRB = offCP + (size_t)P_ * MB * NB * TILE * sizeof(float4);
  size_t offCB = offRB + (size_t)P_ * N_ * sizeof(float4);
  size_t offAC = offCB + (size_t)P_ * M_ * sizeof(float4);

  half_t* nI       = (half_t*)(ws + offNI);
  half_t* nJ       = (half_t*)(ws + offNJ);
  float4* rowPart  = (float4*)(ws + offRP);
  float4* colPart  = (float4*)(ws + offCP);
  float4* rowBest  = (float4*)(ws + offRB);
  float4* colBest  = (float4*)(ws + offCB);
  float*  accum    = (float*) (ws + offAC);

  // Stage A: normalize (one wave per row -> 8 rows per 256-thread block)
  ep_normalize_kernel<<<(P_ * N_) / 8, 256, 0, stream>>>(desc_i, nI, P_ * N_);
  ep_normalize_kernel<<<(P_ * M_) / 8, 256, 0, stream>>>(desc_j, nJ, P_ * M_);
  ep_zero_accum_kernel<<<1, 32, 0, stream>>>(accum);

  // Stage B: WMMA GEMM + tile top-2 partials (in raw dot-product space)
  dim3 grid(MB, NB, P_);
  ep_gemm_top2_kernel<<<grid, 256, 0, stream>>>(nI, nJ, rowPart, colPart);

  // Stage C: merge partials -> final top-2 per row / per column (+ sqrt + ratio)
  ep_merge_top2_kernel<<<(P_ * N_) / 256, 256, 0, stream>>>(rowPart, rowBest, P_ * N_);
  ep_merge_top2_kernel<<<(P_ * M_) / 256, 256, 0, stream>>>(colPart, colBest, P_ * M_);

  // Stage D: mutual match + Sampson, global accumulate
  ep_sampson_kernel<<<(P_ * N_) / 256, 256, 0, stream>>>(rowBest, colBest, kp_i, kp_j, F, accum);

  // Finalize
  ep_finalize_kernel<<<1, 1, 0, stream>>>(accum, (float*)d_out);
}
